// GraphSAGEWithSampling_59880434041042
// MI455X (gfx1250) — matrix operand order repaired and checked by reading, not compiled, not run on hardware
//
#include <hip/hip_runtime.h>
#include <hip/hip_bf16.h>

typedef _Float16 half_t;
typedef __attribute__((ext_vector_type(16))) _Float16 v16h;
typedef __attribute__((ext_vector_type(8)))  float    v8f;

#define N_NODES   100000
#define N_EDGES   1600000
#define DIN       128
#define HID       128
#define OUT_DIM   40
#define OUT_PAD   48

// ---------------------------------------------------------------- utilities
__device__ __forceinline__ float wave_sum(float v) {
  #pragma unroll
  for (int m = 16; m >= 1; m >>= 1) v += __shfl_xor(v, m, 32);
  return v;
}
__device__ __forceinline__ float wave_max(float v) {
  #pragma unroll
  for (int m = 16; m >= 1; m >>= 1) v = fmaxf(v, __shfl_xor(v, m, 32));
  return v;
}

__global__ void k_zero_f32(float* __restrict__ p, int n) {
  int i = blockIdx.x * blockDim.x + threadIdx.x;
  if (i < n) p[i] = 0.0f;
}

// deg accumulation: one thread per edge
__global__ void k_deg(const int* __restrict__ dst, float* __restrict__ deg, int ne) {
  int e = blockIdx.x * blockDim.x + threadIdx.x;
  if (e < ne) atomicAdd(&deg[dst[e]], 1.0f);
}

__global__ void k_invdeg(float* __restrict__ deg, int n) {
  int i = blockIdx.x * blockDim.x + threadIdx.x;
  if (i < n) deg[i] = 1.0f / fmaxf(deg[i], 1.0f);
}

// W [dout x din] f32 -> padded [dout_pad x din] f16 (zero fill past dout)
__global__ void k_cvt_w(const float* __restrict__ w, half_t* __restrict__ wh,
                        int dout, int dout_pad) {
  int i = blockIdx.x * blockDim.x + threadIdx.x;
  int tot = dout_pad * DIN;
  if (i >= tot) return;
  int row = i / DIN;
  wh[i] = (row < dout) ? (half_t)w[i] : (half_t)0.0f;
}

__global__ void k_cvt_f16(const float* __restrict__ in, half_t* __restrict__ outp, int n) {
  int i = blockIdx.x * blockDim.x + threadIdx.x;
  if (i < n) outp[i] = (half_t)in[i];
}

// agg f32 * inv_deg[row] -> f16
__global__ void k_scale_cvt(const float* __restrict__ agg, const float* __restrict__ invd,
                            half_t* __restrict__ outp, int n) {
  int i = blockIdx.x * blockDim.x + threadIdx.x;
  if (i < n) outp[i] = (half_t)(agg[i] * invd[i >> 7]);
}

// ------------------------------------------------ edge scatter (mean-agg numerator)
// one wave per edge; each lane handles 4 features (f32 source: layer 0 reads x)
__global__ void k_agg_f32(const float* __restrict__ h, const int* __restrict__ src,
                          const int* __restrict__ dst, float* __restrict__ agg, int ne) {
  int e    = (blockIdx.x * blockDim.x + threadIdx.x) >> 5;
  int lane = threadIdx.x & 31;
  if (e >= ne) return;
  int s = src[e], d = dst[e];
  const float4 v = *(const float4*)(h + (size_t)s * DIN + lane * 4);
  float* ap = agg + (size_t)d * DIN + lane * 4;
  atomicAdd(ap + 0, v.x); atomicAdd(ap + 1, v.y);
  atomicAdd(ap + 2, v.z); atomicAdd(ap + 3, v.w);
}

// f16 source (layers 1,2 read normalized activations) — halves gather traffic
__global__ void k_agg_f16(const half_t* __restrict__ h, const int* __restrict__ src,
                          const int* __restrict__ dst, float* __restrict__ agg, int ne) {
  int e    = (blockIdx.x * blockDim.x + threadIdx.x) >> 5;
  int lane = threadIdx.x & 31;
  if (e >= ne) return;
  int s = src[e], d = dst[e];
  union { uint2 u; half_t h4[4]; } u;
  u.u = *(const uint2*)(h + (size_t)s * DIN + lane * 4);
  float* ap = agg + (size_t)d * DIN + lane * 4;
  atomicAdd(ap + 0, (float)u.h4[0]); atomicAdd(ap + 1, (float)u.h4[1]);
  atomicAdd(ap + 2, (float)u.h4[2]); atomicAdd(ap + 3, (float)u.h4[3]);
}

// ------------------------------------------------ WMMA GEMM
// out[m,n] = sum_k Ah[m,k]*Wl[n,k] + sum_k Xh[m,k]*Wr[n,k] + bias[n]
// one wave per 16x16 output tile; blockDim = 32 * (ldo/16); grid.x = N/16
__device__ __forceinline__ v16h load_a16(const half_t* base, int k0) {
  union { v16h v; uint4 q[2]; } u;
  u.q[0] = *(const uint4*)(base + k0);
  u.q[1] = *(const uint4*)(base + k0 + 16);
  return u.v;
}
__device__ __forceinline__ v16h load_b16(const half_t* base, int k0) {
  union { v16h v; uint4 q[2]; } u;
  u.q[0] = *(const uint4*)(base + k0);
  u.q[1] = *(const uint4*)(base + k0 + 16);
  return u.v;
}

__global__ void k_gemm_wmma(const half_t* __restrict__ Ah, const half_t* __restrict__ Xh,
                            const half_t* __restrict__ Wl, const half_t* __restrict__ Wr,
                            const float* __restrict__ bias, float* __restrict__ outp,
                            int dout, int ldo) {
  const int lane  = threadIdx.x & 31;
  const int wtile = threadIdx.x >> 5;          // column tile per wave
  const int m0    = blockIdx.x * 16;
  const int n0    = wtile * 16;
  const int hsel  = lane >> 4;                 // 0 / 1
  const int rsub  = lane & 15;

  // A (16x32, MxK): lane<16 -> K {k0..+7, k0+16..+23}; lane>=16 -> K {k0+8..+15, k0+24..+31}
  const half_t* arow = Ah + (size_t)(m0 + rsub) * DIN + 8 * hsel;
  const half_t* xrow = Xh + (size_t)(m0 + rsub) * DIN + 8 * hsel;
  // B (32x16, KxN) from row-major W[n,k]: lane<16 -> K {k0..+7, k0+16..+23}; lane>=16 -> K {k0+8..+15, k0+24..+31}
  const half_t* wlrow = Wl + (size_t)(n0 + rsub) * DIN + 8 * hsel;
  const half_t* wrrow = Wr + (size_t)(n0 + rsub) * DIN + 8 * hsel;

  v8f acc = {};
  #pragma unroll
  for (int k0 = 0; k0 < DIN; k0 += 32) {
    v16h a = load_a16(arow, k0);
    v16h b = load_b16(wlrow, k0);
    acc = __builtin_amdgcn_wmma_f32_16x16x32_f16(false, a, false, b, (short)0, acc,
                                                 false, false);
  }
  #pragma unroll
  for (int k0 = 0; k0 < DIN; k0 += 32) {
    v16h a = load_a16(xrow, k0);
    v16h b = load_b16(wrrow, k0);
    acc = __builtin_amdgcn_wmma_f32_16x16x32_f16(false, a, false, b, (short)0, acc,
                                                 false, false);
  }

  // C/D layout: vgpr r, lanes<16 -> M=r, lanes>=16 -> M=r+8; N = lane&15
  const int col = n0 + rsub;
  const int r0  = m0 + 8 * hsel;
  if (col < dout) {
    const float bv = bias[col];
    #pragma unroll
    for (int r = 0; r < 8; ++r)
      outp[(size_t)(r0 + r) * ldo + col] = acc[r] + bv;
  }
}

// ------------------------------------------------ epilogues
// per-row L2 normalize + relu; write f16 activations for next layer
__global__ void k_norm_relu(const float* __restrict__ in, half_t* __restrict__ hf16, int n) {
  int wave = threadIdx.x >> 5, lane = threadIdx.x & 31;
  int row  = blockIdx.x * (blockDim.x >> 5) + wave;
  if (row >= n) return;
  const float4 v = *(const float4*)(in + (size_t)row * HID + lane * 4);
  float ss = wave_sum(v.x * v.x + v.y * v.y + v.z * v.z + v.w * v.w);
  float inv = 1.0f / fmaxf(sqrtf(ss), 1e-12f);
  union { uint2 u; half_t h4[4]; } o;
  o.h4[0] = (half_t)fmaxf(v.x * inv, 0.0f);
  o.h4[1] = (half_t)fmaxf(v.y * inv, 0.0f);
  o.h4[2] = (half_t)fmaxf(v.z * inv, 0.0f);
  o.h4[3] = (half_t)fmaxf(v.w * inv, 0.0f);
  *(uint2*)(hf16 + (size_t)row * HID + lane * 4) = o.u;
}

// softmax over 40 logits (stored with stride OUT_PAD) -> d_out stride 40
__global__ void k_softmax(const float* __restrict__ in, float* __restrict__ outp, int n) {
  int wave = threadIdx.x >> 5, lane = threadIdx.x & 31;
  int row  = blockIdx.x * (blockDim.x >> 5) + wave;
  if (row >= n) return;
  const float* r = in + (size_t)row * OUT_PAD;
  float v0 = r[lane];
  float v1 = (lane < OUT_DIM - 32) ? r[32 + lane] : -3.0e38f;
  float m  = wave_max(fmaxf(v0, v1));
  float e0 = expf(v0 - m);
  float e1 = (lane < OUT_DIM - 32) ? expf(v1 - m) : 0.0f;
  float s  = wave_sum(e0 + e1);
  float is = 1.0f / s;
  float* o = outp + (size_t)row * OUT_DIM;
  o[lane] = e0 * is;
  if (lane < OUT_DIM - 32) o[32 + lane] = e1 * is;
}

// ---------------------------------------------------------------- launch
extern "C" void kernel_launch(void* const* d_in, const int* in_sizes, int n_in,
                              void* d_out, int out_size, void* d_ws, size_t ws_size,
                              hipStream_t stream) {
  const float* x   = (const float*)d_in[0];
  const int*   ei  = (const int*)d_in[1];
  const float* Wl0 = (const float*)d_in[2];
  const float* Wr0 = (const float*)d_in[3];
  const float* b0  = (const float*)d_in[4];
  const float* Wl1 = (const float*)d_in[5];
  const float* Wr1 = (const float*)d_in[6];
  const float* b1  = (const float*)d_in[7];
  const float* Wl2 = (const float*)d_in[8];
  const float* Wr2 = (const float*)d_in[9];
  const float* b2  = (const float*)d_in[10];
  const int* src = ei;
  const int* dst = ei + N_EDGES;

  // workspace layout (256B aligned)
  char* ws = (char*)d_ws;
  constexpr size_t SZ_INVDEG = (size_t)N_NODES * 4;          //   0.4 MB
  constexpr size_t O_AGG     = (SZ_INVDEG + 255) & ~(size_t)255;
  constexpr size_t SZ_FEAT32 = (size_t)N_NODES * HID * 4;    //  51.2 MB (agg, aliased w/ gemm out)
  constexpr size_t O_HF16    = O_AGG + SZ_FEAT32;
  constexpr size_t SZ_FEAT16 = (size_t)N_NODES * HID * 2;    //  25.6 MB
  constexpr size_t O_AGGF16  = O_HF16 + SZ_FEAT16;
  constexpr size_t O_W       = O_AGGF16 + SZ_FEAT16;
  constexpr size_t SZ_W128   = (size_t)HID * DIN * 2;        //  32 KB
  constexpr size_t SZ_W48    = (size_t)OUT_PAD * DIN * 2;    //  12 KB

  float*  inv_deg = (float*)(ws);
  float*  agg     = (float*)(ws + O_AGG);     // f32 numerator; GEMM output aliases this
  float*  gout    = agg;
  half_t* hf16    = (half_t*)(ws + O_HF16);
  half_t* aggf16  = (half_t*)(ws + O_AGGF16);
  half_t* wl0h = (half_t*)(ws + O_W);
  half_t* wr0h = (half_t*)(ws + O_W + SZ_W128);
  half_t* wl1h = (half_t*)(ws + O_W + 2 * SZ_W128);
  half_t* wr1h = (half_t*)(ws + O_W + 3 * SZ_W128);
  half_t* wl2h = (half_t*)(ws + O_W + 4 * SZ_W128);
  half_t* wr2h = (half_t*)(ws + O_W + 4 * SZ_W128 + SZ_W48);

  const int feat_n   = N_NODES * HID;                  // 12.8M
  const int g_feat   = (feat_n + 255) / 256;           // 50000
  const int g_nodes  = (N_NODES + 255) / 256;
  const int g_edges  = (N_EDGES + 255) / 256;          // 6250
  const int g_edgew  = N_EDGES / 8;                    // 200000 (8 edge-waves / 256-thr block)
  const int g_rows8  = (N_NODES + 7) / 8;              // 12500
  const int g_gemm   = N_NODES / 16;                   // 6250

  // ---- degrees
  k_zero_f32<<<g_nodes, 256, 0, stream>>>(inv_deg, N_NODES);
  k_deg<<<g_edges, 256, 0, stream>>>(dst, inv_deg, N_EDGES);
  k_invdeg<<<g_nodes, 256, 0, stream>>>(inv_deg, N_NODES);

  // ---- weights -> f16 (padded)
  k_cvt_w<<<64, 256, 0, stream>>>(Wl0, wl0h, HID, HID);
  k_cvt_w<<<64, 256, 0, stream>>>(Wr0, wr0h, HID, HID);
  k_cvt_w<<<64, 256, 0, stream>>>(Wl1, wl1h, HID, HID);
  k_cvt_w<<<64, 256, 0, stream>>>(Wr1, wr1h, HID, HID);
  k_cvt_w<<<24, 256, 0, stream>>>(Wl2, wl2h, OUT_DIM, OUT_PAD);
  k_cvt_w<<<24, 256, 0, stream>>>(Wr2, wr2h, OUT_DIM, OUT_PAD);

  // ---- layer 0 (input x is f32)
  k_cvt_f16<<<g_feat, 256, 0, stream>>>(x, hf16, feat_n);
  k_zero_f32<<<g_feat, 256, 0, stream>>>(agg, feat_n);
  k_agg_f32<<<g_edgew, 256, 0, stream>>>(x, src, dst, agg, N_EDGES);
  k_scale_cvt<<<g_feat, 256, 0, stream>>>(agg, inv_deg, aggf16, feat_n);
  k_gemm_wmma<<<g_gemm, 256, 0, stream>>>(aggf16, hf16, wl0h, wr0h, b0, gout, HID, HID);
  k_norm_relu<<<g_rows8, 256, 0, stream>>>(gout, hf16, N_NODES);

  // ---- layer 1
  k_zero_f32<<<g_feat, 256, 0, stream>>>(agg, feat_n);
  k_agg_f16<<<g_edgew, 256, 0, stream>>>(hf16, src, dst, agg, N_EDGES);
  k_scale_cvt<<<g_feat, 256, 0, stream>>>(agg, inv_deg, aggf16, feat_n);
  k_gemm_wmma<<<g_gemm, 256, 0, stream>>>(aggf16, hf16, wl1h, wr1h, b1, gout, HID, HID);
  k_norm_relu<<<g_rows8, 256, 0, stream>>>(gout, hf16, N_NODES);

  // ---- layer 2 (dout=40, padded to 48; 3 waves per block)
  k_zero_f32<<<g_feat, 256, 0, stream>>>(agg, feat_n);
  k_agg_f16<<<g_edgew, 256, 0, stream>>>(hf16, src, dst, agg, N_EDGES);
  k_scale_cvt<<<g_feat, 256, 0, stream>>>(agg, inv_deg, aggf16, feat_n);
  k_gemm_wmma<<<g_gemm, 96, 0, stream>>>(aggf16, hf16, wl2h, wr2h, b2, gout, OUT_DIM, OUT_PAD);
  k_softmax<<<g_rows8, 256, 0, stream>>>(gout, (float*)d_out, N_NODES);
}
